// AxialSelfAttention2d_23450521436349
// MI455X (gfx1250) — compile-verified, matrix operand order checked by reading, not run on hardware
//
#include <hip/hip_runtime.h>
#include <hip/hip_bf16.h>

// ---------------- CDNA5 WMMA types / helpers ----------------
typedef __attribute__((ext_vector_type(16))) __bf16 v16bf;
typedef __attribute__((ext_vector_type(2)))  __bf16 v2bf;
typedef __attribute__((ext_vector_type(8)))  float  v8f;
typedef __attribute__((ext_vector_type(4)))  int    v4i;

#define PIX 65536   // S*L
#define SS  256
#define DH  64

#if __has_builtin(__builtin_amdgcn_global_load_async_to_lds_b128) && \
    __has_builtin(__builtin_amdgcn_s_wait_asynccnt)
#define USE_ASYNC 1
// builtin signature (probe-confirmed): (v4i AS1* src, v4i AS3* dst, imm int, imm int)
typedef __attribute__((address_space(1))) v4i gv4i;
typedef __attribute__((address_space(3))) v4i lv4i;
#else
#define USE_ASYNC 0
#endif

// Native casts: backend fuses into v_cvt_pk_bf16_f32 on gfx1250 (verified in asm).
static __device__ __forceinline__ unsigned short f2bf(float f) {
  union { __bf16 b; unsigned short u; } c;
  c.b = (__bf16)f;
  return c.u;
}
static __device__ __forceinline__ unsigned int packbf(float a, float b) {
  union { v2bf v; unsigned int u; } c;
  c.v.x = (__bf16)a;
  c.v.y = (__bf16)b;
  return c.u;
}
static __device__ __forceinline__ float bf2f(unsigned short h) {
  return __uint_as_float(((unsigned int)h) << 16);
}
static __device__ __forceinline__ v8f vzero8() {
  v8f z = {0.f, 0.f, 0.f, 0.f, 0.f, 0.f, 0.f, 0.f};
  return z;
}
// Load a 16xK=32 bf16 WMMA A/B fragment from a row-major LDS tile.
// Per ISA layout: lane%16 selects row; VGPR v holds K pair
// k = (v<4?0:16) + (lane/16)*8 + (v%4)*2.
static __device__ __forceinline__ v16bf ldsfrag(const unsigned short* base,
                                                int row, int ld, int kc, int hi) {
  union { v16bf v; unsigned int u[8]; } f;
  const unsigned short* p = base + row * ld + kc + hi * 8;
#pragma unroll
  for (int r = 0; r < 8; ++r) {
    int k = ((r & 4) << 2) + ((r & 3) << 1);
    f.u[r] = *(const unsigned int*)(p + k);
  }
  return f.v;
}
static __device__ __forceinline__ v8f wmma_bf16(v16bf a, v16bf b, v8f c) {
  return __builtin_amdgcn_wmma_f32_16x16x32_bf16(false, a, false, b,
                                                 (short)0, c, false, false);
}

// ---------------- QKV projection GEMM ----------------
// Y[1536][PIX] (bf16) = W[1536][512] * X[512][PIX] + bias, f32 inputs,
// bf16 WMMA / f32 accumulate. Block 128x128, 8 waves (wave = 32x64),
// double-buffered LDS + register-staged global prefetch pipeline.
__global__ __launch_bounds__(256) void qkv_gemm(
    const float* __restrict__ W, const float* __restrict__ bias,
    const float* __restrict__ X, unsigned short* __restrict__ Y) {
  __shared__ __align__(16) unsigned short sm[17408];  // 34 KB
  // buffer b: As = sm + b*8704 ([128][34]), Bs = As + 4352 ([128 n][34 k])
  unsigned short* Cs = sm;  // [128][136] bf16, reused after K loop

  const int t = threadIdx.x;
  const int bm = blockIdx.y * 128;
  const int bn = blockIdx.x * 128;
  const int lane = t & 31, wave = t >> 5;
  const int hi = lane >> 4, ln = lane & 15;
  const int wm = (wave & 3) * 32;    // 4 waves along M
  const int wn = (wave >> 2) * 64;   // 2 waves along N
  const int r0 = t >> 3, c4 = (t & 7) << 2;   // A staging map
  const int k0 = t >> 5, n4 = (t & 31) << 2;  // B staging map

  v8f acc[2][4];
#pragma unroll
  for (int i = 0; i < 2; ++i)
#pragma unroll
    for (int j = 0; j < 4; ++j) acc[i][j] = vzero8();

  float4 ra[4], rb[4];
#define LOAD_GLOBAL(KC)                                                        \
  {                                                                            \
    _Pragma("unroll") for (int rr = 0; rr < 4; ++rr) {                         \
      ra[rr] = *(const float4*)(W + (size_t)(bm + r0 + rr * 32) * 512 + (KC) + c4); \
      rb[rr] = *(const float4*)(X + (size_t)((KC) + k0 + rr * 8) * PIX + bn + n4);  \
    }                                                                          \
  }
#define STORE_LDS(BUF)                                                         \
  {                                                                            \
    unsigned short* Asb = sm + (BUF) * 8704;                                   \
    unsigned short* Bsb = Asb + 4352;                                          \
    _Pragma("unroll") for (int rr = 0; rr < 4; ++rr) {                         \
      int row = r0 + rr * 32;                                                  \
      *(unsigned int*)(Asb + row * 34 + c4)     = packbf(ra[rr].x, ra[rr].y);  \
      *(unsigned int*)(Asb + row * 34 + c4 + 2) = packbf(ra[rr].z, ra[rr].w);  \
      int k = k0 + rr * 8;                                                     \
      Bsb[(n4 + 0) * 34 + k] = f2bf(rb[rr].x);                                 \
      Bsb[(n4 + 1) * 34 + k] = f2bf(rb[rr].y);                                 \
      Bsb[(n4 + 2) * 34 + k] = f2bf(rb[rr].z);                                 \
      Bsb[(n4 + 3) * 34 + k] = f2bf(rb[rr].w);                                 \
    }                                                                          \
  }

  LOAD_GLOBAL(0);
  STORE_LDS(0);
  __syncthreads();
  for (int kk = 0; kk < 16; ++kk) {
    if (kk < 15) LOAD_GLOBAL((kk + 1) * 32);  // prefetch next chunk -> regs
    unsigned short* Asb = sm + (kk & 1) * 8704;
    unsigned short* Bsb = Asb + 4352;
    // all fragment ds_loads issued before the 8 WMMAs
    v16bf a0 = ldsfrag(Asb, wm + ln, 34, 0, hi);
    v16bf a1 = ldsfrag(Asb, wm + 16 + ln, 34, 0, hi);
    v16bf b[4];
#pragma unroll
    for (int nt = 0; nt < 4; ++nt)
      b[nt] = ldsfrag(Bsb, wn + nt * 16 + ln, 34, 0, hi);
#pragma unroll
    for (int nt = 0; nt < 4; ++nt) {
      acc[0][nt] = wmma_bf16(a0, b[nt], acc[0][nt]);
      acc[1][nt] = wmma_bf16(a1, b[nt], acc[1][nt]);
    }
    if (kk < 15) STORE_LDS((kk + 1) & 1);  // fill other buffer
    __syncthreads();
  }
#undef LOAD_GLOBAL
#undef STORE_LDS

  // restage C tile in LDS so the global bf16 stores are fully coalesced
#pragma unroll
  for (int mt = 0; mt < 2; ++mt)
#pragma unroll
    for (int nt = 0; nt < 4; ++nt)
#pragma unroll
      for (int v = 0; v < 8; ++v)
        Cs[(wm + mt * 16 + v + 8 * hi) * 136 + wn + nt * 16 + ln] =
            f2bf(acc[mt][nt][v]);
  __syncthreads();
  const int r = t >> 1, cb = (t & 1) << 6;
  const float bv = bias[bm + r];
  unsigned short* yrow = Y + (size_t)(bm + r) * PIX + bn + cb;
  const unsigned short* crow = Cs + r * 136 + cb;
#pragma unroll
  for (int j = 0; j < 64; j += 2)
    *(unsigned int*)(yrow + j) = packbf(bf2f(crow[j]) + bv, bf2f(crow[j + 1]) + bv);
}

// ---------------- Fused axial attention ----------------
// One workgroup per (head h, fixed coordinate f). STRIDE=1: row attention
// (attend along L). STRIDE=256: column attention (attend along S).
template <int STRIDE>
__global__ __launch_bounds__(256) void attn_axial(
    const unsigned short* __restrict__ qkv,  // bf16 [1536][PIX]
    const float* __restrict__ src,           // residual input  [512][PIX]
    float* __restrict__ dst) {               // output          [512][PIX]
  // LDS (shorts): Kb[256][68] | Vb[64][264] | Qb[128][68] | Pb[128][258] | inv[128]f32
  // Ob (f32 [64][258]) overlays Pb after all P reads are done. ~152.6 KB.
  __shared__ __align__(16) unsigned short sm[76288];
  unsigned short* Kb = sm;                  // K, [j][c] (K-dim contiguous)
  unsigned short* Vb = sm + 17408;          // V, [d][j] (ld 264: 16B rows)
  unsigned short* Qb = sm + 34304;          // Q half, [i][c]
  unsigned short* Pb = sm + 43008;          // probs,  [i][j]
  float* Ob = (float*)(sm + 43008);         // out tile, [d][i] (overlay)
  float* rowinv = (float*)(sm + 76032);     // 1/rowsum

  const int t = threadIdx.x;
  const int f = blockIdx.x;
  const int h = blockIdx.y;
  const int fixedOff = (STRIDE == 1) ? f * SS : f;
  const int lane = t & 31, wave = t >> 5;
  const int hi = lane >> 4, ln = lane & 15;

  // phase 0: stage K (transposed) and V for all 256 positions
  {
    const int c = t >> 2, j0 = (t & 3) << 6;
    const unsigned short* kr = qkv + (size_t)(512 + h * 64 + c) * PIX + fixedOff;
    const unsigned short* vr = qkv + (size_t)(1024 + h * 64 + c) * PIX + fixedOff;
    bool asyncV = false;
#if USE_ASYNC
    if constexpr (STRIDE == 1) {
      // V rows are contiguous in global and in LDS: DMA straight to LDS
      // on the ASYNCcnt path, overlapping with the scalar K staging below.
      asyncV = true;
#pragma unroll
      for (int jj = 0; jj < 64; jj += 8)
        __builtin_amdgcn_global_load_async_to_lds_b128(
            (gv4i*)(vr + j0 + jj),
            (lv4i*)(Vb + c * 264 + j0 + jj),
            0, 0);
    }
#endif
#pragma unroll 4
    for (int jj = 0; jj < 64; ++jj) {
      int j = j0 + jj;
      Kb[j * 68 + c] = kr[(size_t)j * STRIDE];
    }
    if (!asyncV) {
#pragma unroll 4
      for (int jj = 0; jj < 64; ++jj) {
        int j = j0 + jj;
        Vb[c * 264 + j] = vr[(size_t)j * STRIDE];
      }
    }
#if USE_ASYNC
    if constexpr (STRIDE == 1) __builtin_amdgcn_s_wait_asynccnt(0);
#endif
  }

  for (int half = 0; half < 2; ++half) {
    const int iBase = half * 128;
    // phase 1: stage this half's 128 query rows
    {
      const int c = t >> 2, i0 = (t & 3) << 5;
      const unsigned short* qr = qkv + (size_t)(h * 64 + c) * PIX + fixedOff;
#pragma unroll 4
      for (int ii = 0; ii < 32; ++ii) {
        int i = i0 + ii;
        Qb[i * 68 + c] = qr[(size_t)(iBase + i) * STRIDE];
      }
    }
    __syncthreads();  // staging visible; prev half's Ob reads complete

    // phase 2: S = Q^T K  (128x256, K-dim 64) -> raw bf16 scores in Pb
    {
      const int m = wave * 16;  // 16 rows per wave
      v16bf aq0 = ldsfrag(Qb, m + ln, 68, 0, hi);
      v16bf aq1 = ldsfrag(Qb, m + ln, 68, 32, hi);
#pragma unroll
      for (int nt = 0; nt < 16; ++nt) {
        v16bf bk0 = ldsfrag(Kb, nt * 16 + ln, 68, 0, hi);
        v16bf bk1 = ldsfrag(Kb, nt * 16 + ln, 68, 32, hi);
        v8f s = vzero8();
        s = wmma_bf16(aq0, bk0, s);
        s = wmma_bf16(aq1, bk1, s);
#pragma unroll
        for (int v = 0; v < 8; ++v)
          Pb[(m + v + 8 * hi) * 258 + nt * 16 + ln] = f2bf(s[v]);
      }
    }
    __syncthreads();

    // phase 3: softmax over j (unnormalized exp kept in P, 1/sum in rowinv)
    if (t < 128) {
      unsigned short* pr = Pb + t * 258;
      float mx = -3.0e38f;
#pragma unroll 8
      for (int j = 0; j < 256; ++j) mx = fmaxf(mx, bf2f(pr[j]));
      float sum = 0.f;
#pragma unroll 8
      for (int j = 0; j < 256; ++j) {
        float e = __expf(bf2f(pr[j]) - mx);
        sum += e;
        pr[j] = f2bf(e);
      }
      rowinv[t] = 1.f / sum;
    }
    __syncthreads();

    // phase 4: O = exp(S) * V^T  (128x64, K-dim 256)
    v8f o[4];
#pragma unroll
    for (int nt = 0; nt < 4; ++nt) o[nt] = vzero8();
    const int m = wave * 16;
#pragma unroll
    for (int kc = 0; kc < 256; kc += 32) {
      v16bf ap = ldsfrag(Pb, m + ln, 258, kc, hi);
#pragma unroll
      for (int nt = 0; nt < 4; ++nt) {
        v16bf bv = ldsfrag(Vb, nt * 16 + ln, 264, kc, hi);
        o[nt] = wmma_bf16(ap, bv, o[nt]);
      }
    }
    __syncthreads();  // everyone done reading Pb before Ob overlays it

    // phase 5: normalize (fold softmax 1/sum) and stash O as [d][i] f32
    float rv[8];
#pragma unroll
    for (int v = 0; v < 8; ++v) rv[v] = rowinv[m + v + 8 * hi];
#pragma unroll
    for (int nt = 0; nt < 4; ++nt)
#pragma unroll
      for (int v = 0; v < 8; ++v)
        Ob[(nt * 16 + ln) * 258 + m + v + 8 * hi] = o[nt][v] * rv[v];
    __syncthreads();

    // phase 6: residual add + write out (channel-major, i contiguous)
    {
      const int d = t >> 2, i0 = (t & 3) << 5;
      const size_t ch = (size_t)(h * 64 + d);
      const float* s = src + ch * PIX + fixedOff;
      float* dp = dst + ch * PIX + fixedOff;
      const float* orow = Ob + d * 258;
#pragma unroll 4
      for (int ii = 0; ii < 32; ++ii) {
        int i = i0 + ii;
        size_t gi = (size_t)(iBase + i) * STRIDE;
        dp[gi] = s[gi] + orow[i];
      }
    }
    // next half's phase-1 barrier protects Pb/Ob reuse
  }
}

// ---------------- launch ----------------
extern "C" void kernel_launch(void* const* d_in, const int* in_sizes, int n_in,
                              void* d_out, int out_size, void* d_ws, size_t ws_size,
                              hipStream_t stream) {
  const float* x  = (const float*)d_in[0];   // [512][256][256]
  const float* Wr = (const float*)d_in[1];   // [1536][512]
  const float* br = (const float*)d_in[2];   // [1536]
  const float* Wc = (const float*)d_in[3];
  const float* bc = (const float*)d_in[4];
  float* out = (float*)d_out;                // [512][256][256]
  // qkv intermediate, bf16 [1536][65536] = 192 MB, reused across both passes
  unsigned short* qkvws = (unsigned short*)d_ws;

  dim3 gg(PIX / 128, 1536 / 128);            // 512 x 12 blocks
  dim3 ga(SS, 8);                            // (fixed coord) x heads

  // pass 1: row attention
  qkv_gemm<<<gg, 256, 0, stream>>>(Wr, br, x, qkvws);
  attn_axial<1><<<ga, 256, 0, stream>>>(qkvws, x, out);
  // pass 2: column attention (reads/writes `out` in place, disjoint per block)
  qkv_gemm<<<gg, 256, 0, stream>>>(Wc, bc, out, qkvws);
  attn_axial<SS><<<ga, 256, 0, stream>>>(qkvws, out, out);
}
